// MultiHeadAttn_1786706395825
// MI455X (gfx1250) — matrix operand order repaired and checked by reading, not compiled, not run on hardware
//
#include <hip/hip_runtime.h>
#include <stdint.h>

// ---------------------------------------------------------------------------
// Problem constants (B=4, S=2048, D=1024, H=16, Dh=64)
// ---------------------------------------------------------------------------
#define BB   4
#define SS   2048
#define DD   1024
#define HH_  16
#define DH   64
#define NEGV (-1.0e30f)

typedef __attribute__((ext_vector_type(16))) __bf16 v16bf;
typedef __attribute__((ext_vector_type(8)))  float  v8f;
typedef __attribute__((ext_vector_type(4)))  int    v4i;

union FragBF {
    v16bf bf;
    uint4 u4[2];
};

static __device__ __forceinline__ v8f v8f_zero() {
    v8f z;
#pragma unroll
    for (int i = 0; i < 8; ++i) z[i] = 0.0f;
    return z;
}

static __device__ __forceinline__ v8f wmma_bf16(const v16bf a, const v16bf b, v8f c) {
    return __builtin_amdgcn_wmma_f32_16x16x32_bf16(false, a, false, b, (short)0, c,
                                                   false, false);
}

static __device__ __forceinline__ unsigned short f2bf(float f) {
    unsigned int u = __float_as_uint(f);
    u += 0x7FFFu + ((u >> 16) & 1u);   // round-to-nearest-even
    return (unsigned short)(u >> 16);
}

// ---------------------------------------------------------------------------
// Async global->LDS staging (CDNA5 ASYNCcnt path), with safe fallback.
// Builtin wants typed int4 pointers: AS1 (global, 64-bit) and AS3 (LDS,
// 32-bit).  Casts go through integers (AS3 ptr value == LDS offset == low 32
// bits of the generic shared pointer).
// ---------------------------------------------------------------------------
#if defined(__has_builtin)
#if __has_builtin(__builtin_amdgcn_global_load_async_to_lds_b128)
#define HAS_ASYNC_LDS 1
#endif
#endif

typedef __attribute__((address_space(1))) v4i gas_v4i;
typedef __attribute__((address_space(3))) v4i las_v4i;

#ifdef HAS_ASYNC_LDS
#define CPY16(l, g)                                                           \
    __builtin_amdgcn_global_load_async_to_lds_b128(                           \
        (gas_v4i*)(uintptr_t)(g),                                             \
        (las_v4i*)(uint32_t)(uintptr_t)(l), 0, 0)
#if __has_builtin(__builtin_amdgcn_s_wait_asynccnt)
#define WAIT_ASYNC(n) __builtin_amdgcn_s_wait_asynccnt(n)
#else
#define WAIT_ASYNC(n) asm volatile("s_wait_asynccnt %0" ::"i"(n) : "memory")
#endif
#else
#define CPY16(l, g) (*(uint4*)(l) = *(const uint4*)(g))
#define WAIT_ASYNC(n) ((void)0)
#endif

// ---------------------------------------------------------------------------
// Kernel 0: fp32 -> bf16 convert
// ---------------------------------------------------------------------------
__global__ void k_cvt_bf16(const float* __restrict__ in,
                           unsigned short* __restrict__ out, int n) {
    int i = blockIdx.x * blockDim.x + threadIdx.x;
    if (i < n) out[i] = f2bf(in[i]);
}

// ---------------------------------------------------------------------------
// Shared GEMM core: block = 128 thr (4 waves), block tile 128x128,
// wave tile 64x64 (4x4 WMMA), K staged through double-buffered LDS
// via async-to-LDS.  A[M,K] bf16 row-major, W[N,K] bf16 row-major.
// ---------------------------------------------------------------------------
#define GEMM_CORE(XPTR, WPTR)                                                 \
    __shared__ __align__(16) unsigned short aS[2][128 * 32];                  \
    __shared__ __align__(16) unsigned short bS[2][128 * 32];                  \
    const int lane = threadIdx.x & 31;                                        \
    const int wav  = threadIdx.x >> 5;                                        \
    const int wr   = wav >> 1;                                                \
    const int wc   = wav & 1;                                                 \
    const int rowBlk = blockIdx.y * 128;                                      \
    const int colBlk = blockIdx.x * 128;                                      \
    const int rowBase = rowBlk + wr * 64;                                     \
    const int colBase = colBlk + wc * 64;                                     \
    const int ml = lane & 15;                                                 \
    const int hh = lane >> 4;                                                 \
    v8f acc[4][4];                                                            \
    _Pragma("unroll") for (int i = 0; i < 4; ++i)                             \
        _Pragma("unroll") for (int j = 0; j < 4; ++j) acc[i][j] = v8f_zero(); \
    auto stage = [&](int kk, int bi) {                                        \
        const int t = threadIdx.x;                                            \
        _Pragma("unroll") for (int c = 0; c < 4; ++c) {                       \
            const size_t o   = (size_t)(t + c * 128) * 16;                    \
            const int    row = (int)(o >> 6);                                 \
            const int    col = (int)(o & 63);                                 \
            CPY16((char*)&aS[bi][0] + o,                                      \
                  (const char*)(XPTR) + ((size_t)(rowBlk + row)) * 2048 +     \
                      (size_t)kk * 2 + col);                                  \
            CPY16((char*)&bS[bi][0] + o,                                      \
                  (const char*)(WPTR) + ((size_t)(colBlk + row)) * 2048 +     \
                      (size_t)kk * 2 + col);                                  \
        }                                                                     \
    };                                                                        \
    stage(0, 0);                                                              \
    for (int s = 0; s < 32; ++s) {                                            \
        const int bi = s & 1;                                                 \
        if (s + 1 < 32) {                                                     \
            stage((s + 1) * 32, bi ^ 1);                                      \
            WAIT_ASYNC(8);                                                    \
        } else {                                                              \
            WAIT_ASYNC(0);                                                    \
        }                                                                     \
        __syncthreads();                                                      \
        FragBF af[4];                                                         \
        _Pragma("unroll") for (int i = 0; i < 4; ++i) {                       \
            const unsigned short* p = &aS[bi][0] + (wr * 64 + i * 16 + ml) * 32; \
            af[i].u4[0] = *(const uint4*)(p + hh * 8);                        \
            af[i].u4[1] = *(const uint4*)(p + 16 + hh * 8);                   \
        }                                                                     \
        _Pragma("unroll") for (int j = 0; j < 4; ++j) {                       \
            const unsigned short* p = &bS[bi][0] + (wc * 64 + j * 16 + ml) * 32; \
            FragBF bfj;                                                       \
            bfj.u4[0] = *(const uint4*)(p + hh * 8);                          \
            bfj.u4[1] = *(const uint4*)(p + 16 + hh * 8);                     \
            _Pragma("unroll") for (int i = 0; i < 4; ++i)                     \
                acc[i][j] = wmma_bf16(af[i].bf, bfj.bf, acc[i][j]);           \
        }                                                                     \
        __syncthreads();                                                      \
    }

// ---------------------------------------------------------------------------
// Kernel 1: QKV GEMM.  C[8192,3072] = Xbf * Wqkv^T, scattered into
// q[B,H,S,Dh], k[B,H,S,Dh], vT[B,H,Dh,S] (bf16).
// ---------------------------------------------------------------------------
__global__ void __launch_bounds__(128)
k_qkv_gemm(const unsigned short* __restrict__ X,
           const unsigned short* __restrict__ W,
           unsigned short* __restrict__ qb,
           unsigned short* __restrict__ kb,
           unsigned short* __restrict__ vT) {
    GEMM_CORE(X, W)

#pragma unroll
    for (int i = 0; i < 4; ++i) {
#pragma unroll
        for (int j = 0; j < 4; ++j) {
            const int col  = colBase + j * 16 + ml;   // 0..3071
            const int part = col >> 10;               // 0=q 1=k 2=v
            const int d    = col & (DD - 1);
            const int head = d >> 6;
            const int dh   = d & 63;
#pragma unroll
            for (int r = 0; r < 8; ++r) {
                const int rr = rowBase + i * 16 + hh * 8 + r;   // global M row
                const int b  = rr >> 11;
                const int s  = rr & (SS - 1);
                const unsigned short h16 = f2bf(acc[i][j][r]);
                if (part == 0)
                    qb[(((size_t)(b * HH_ + head) * SS) + s) * DH + dh] = h16;
                else if (part == 1)
                    kb[(((size_t)(b * HH_ + head) * SS) + s) * DH + dh] = h16;
                else
                    vT[(((size_t)(b * HH_ + head) * DH) + dh) * SS + s] = h16;
            }
        }
    }
}

// ---------------------------------------------------------------------------
// Kernel 3: FC GEMM.  out[8192,1024] = ctx * Wfc^T + b_fc (fp32 out)
// ---------------------------------------------------------------------------
__global__ void __launch_bounds__(128)
k_fc_gemm(const unsigned short* __restrict__ A,
          const unsigned short* __restrict__ W,
          const float* __restrict__ bfc,
          float* __restrict__ out) {
    GEMM_CORE(A, W)

#pragma unroll
    for (int j = 0; j < 4; ++j) {
        const int col = colBase + j * 16 + ml;
        const float bias = bfc[col];
#pragma unroll
        for (int i = 0; i < 4; ++i) {
#pragma unroll
            for (int r = 0; r < 8; ++r) {
                const int rr = rowBase + i * 16 + hh * 8 + r;
                out[(size_t)rr * DD + col] = acc[i][j][r] + bias;
            }
        }
    }
}

// ---------------------------------------------------------------------------
// Kernel 2: flash attention.  256 thr = 8 waves, all 8 waves of a block share
// one (b,h); each wave owns one 16-row q-tile.  K/V chunks (32 keys) staged
// in double-buffered LDS via async-to-LDS, shared by all waves (8x reuse).
// Probabilities staged through LDS (C-layout -> A-layout).
// ---------------------------------------------------------------------------
__global__ void __launch_bounds__(256)
k_attention(const unsigned short* __restrict__ Q,    // [B*H, S, 64]
            const unsigned short* __restrict__ Kk,   // [B*H, S, 64]
            const unsigned short* __restrict__ VT,   // [B*H, 64, S]
            const int* __restrict__ mask,            // [B, S]
            const int* __restrict__ useg,            // [1]
            const float* __restrict__ shiftp,        // [1]
            const float* __restrict__ biaspp,        // [1]
            unsigned short* __restrict__ ctx) {      // [B, S, D] bf16
    __shared__ __align__(16) unsigned short kS[2][32 * 64];   // 4KB x2
    __shared__ __align__(16) unsigned short vS[2][64 * 32];   // 4KB x2
    __shared__ __align__(16) unsigned short pS[8][16 * 32];   // 1KB/wave

    const int lane = threadIdx.x & 31;
    const int wav  = threadIdx.x >> 5;
    const int nl   = lane & 15;
    const int hh   = lane >> 4;

    const int tile = blockIdx.x * 8 + wav;   // 0..8191
    const int bh   = tile >> 7;              // same for all waves in block
    const int st   = tile & 127;
    const int b    = bh >> 4;
    const int head = bh & 15;

    const int   ug     = useg[0];
    const float shiftv = shiftp[0];
    const float biasv  = biaspp[0];

    // Q A-fragments (Dh 0..31 / 32..63), loaded once from global
    const unsigned short* Qp = Q + ((size_t)bh * SS + st * 16 + nl) * DH;
    FragBF qf0, qf1;
    qf0.u4[0] = *(const uint4*)(Qp + hh * 8);
    qf0.u4[1] = *(const uint4*)(Qp + 16 + hh * 8);
    qf1.u4[0] = *(const uint4*)(Qp + 32 + hh * 8);
    qf1.u4[1] = *(const uint4*)(Qp + 48 + hh * 8);

    v8f o[4];
#pragma unroll
    for (int j = 0; j < 4; ++j) o[j] = v8f_zero();
    float mrow[8], lrow[8];
#pragma unroll
    for (int r = 0; r < 8; ++r) { mrow[r] = -3.0e38f; lrow[r] = 0.0f; }

    unsigned short* myp = &pS[wav][0];
    const int* mrowp = mask + b * SS;

    // 256 threads x 16B == one 4KB tile each for K and V
    auto stage_kv = [&](int kcol, int bi) {
        const size_t o16 = (size_t)threadIdx.x * 16;
        {   // K chunk: 32 rows x 128B
            const int row = (int)(o16 >> 7);
            const int col = (int)(o16 & 127);
            CPY16((char*)&kS[bi][0] + o16,
                  (const char*)Kk + ((size_t)bh * SS + kcol + row) * 128 + col);
        }
        {   // V^T chunk: 64 rows x 64B
            const int row = (int)(o16 >> 6);
            const int col = (int)(o16 & 63);
            CPY16((char*)&vS[bi][0] + o16,
                  (const char*)VT + ((size_t)bh * DH + row) * (SS * 2) +
                      (size_t)kcol * 2 + col);
        }
    };

    stage_kv(0, 0);
    for (int c = 0; c < SS / 32; ++c) {
        const int bi  = c & 1;
        const int kbx = c * 32;
        if (c + 1 < SS / 32) {
            stage_kv(kbx + 32, bi ^ 1);
            WAIT_ASYNC(2);
        } else {
            WAIT_ASYNC(0);
        }
        __syncthreads();   // chunk c visible to all waves

        // ---- scores: two 16x16 tiles over Dh=64 (2 WMMA each), K from LDS ----
        const unsigned short* kp0 = &kS[bi][0] + nl * 64;
        const unsigned short* kp1 = &kS[bi][0] + (16 + nl) * 64;
        FragBF kfl, kfh;
        v8f s0 = v8f_zero(), s1 = v8f_zero();

        kfl.u4[0] = *(const uint4*)(kp0 + hh * 8);
        kfl.u4[1] = *(const uint4*)(kp0 + 16 + hh * 8);
        kfh.u4[0] = *(const uint4*)(kp0 + 32 + hh * 8);
        kfh.u4[1] = *(const uint4*)(kp0 + 48 + hh * 8);
        s0 = wmma_bf16(qf0.bf, kfl.bf, s0);
        s0 = wmma_bf16(qf1.bf, kfh.bf, s0);

        kfl.u4[0] = *(const uint4*)(kp1 + hh * 8);
        kfl.u4[1] = *(const uint4*)(kp1 + 16 + hh * 8);
        kfh.u4[0] = *(const uint4*)(kp1 + 32 + hh * 8);
        kfh.u4[1] = *(const uint4*)(kp1 + 48 + hh * 8);
        s1 = wmma_bf16(qf0.bf, kfl.bf, s1);
        s1 = wmma_bf16(qf1.bf, kfh.bf, s1);

        // ---- scale, key-mask, Gaussian positional bias ----
        const int j0 = kbx + nl;
        const int j1 = kbx + 16 + nl;
        const int km0 = mrowp[j0];
        const int km1 = mrowp[j1];
        float sc0[8], sc1[8];
#pragma unroll
        for (int r = 0; r < 8; ++r) {
            const int irow = st * 16 + hh * 8 + r;
            float v0 = s0[r] * 0.125f;           // / sqrt(64)
            float v1 = s1[r] * 0.125f;
            if (km0 == 0) v0 = NEGV;
            if (km1 == 0) v1 = NEGV;
            if (ug) {
                float d0 = (float)(irow - j0);
                float d1 = (float)(irow - j1);
                v0 -= shiftv * d0 * d0 + biasv;
                v1 -= shiftv * d1 * d1 + biasv;
            }
            sc0[r] = v0; sc1[r] = v1;
        }

        // ---- online softmax (row lives in a 16-lane group) ----
        float corr[8];
#pragma unroll
        for (int r = 0; r < 8; ++r) {
            float tmax = fmaxf(sc0[r], sc1[r]);
#pragma unroll
            for (int off = 1; off < 16; off <<= 1)
                tmax = fmaxf(tmax, __shfl_xor(tmax, off, 32));
            const float nm = fmaxf(mrow[r], tmax);
            const float cr = __expf(mrow[r] - nm);
            mrow[r] = nm;
            corr[r] = cr;
            const float p0 = __expf(sc0[r] - nm);
            const float p1 = __expf(sc1[r] - nm);
            float ps = p0 + p1;
#pragma unroll
            for (int off = 1; off < 16; off <<= 1)
                ps += __shfl_xor(ps, off, 32);
            lrow[r] = lrow[r] * cr + ps;
            myp[(hh * 8 + r) * 32 + nl]      = f2bf(p0);
            myp[(hh * 8 + r) * 32 + 16 + nl] = f2bf(p1);
        }
#pragma unroll
        for (int j = 0; j < 4; ++j)
#pragma unroll
            for (int r = 0; r < 8; ++r) o[j][r] *= corr[r];

        __syncthreads();   // P store -> P load (uniform across block)

        // ---- P A-fragment from LDS (16x32) ----
        FragBF pf;
        pf.u4[0] = *(const uint4*)(myp + nl * 32 + hh * 8);
        pf.u4[1] = *(const uint4*)(myp + nl * 32 + 16 + hh * 8);

        // ---- PV: 4 WMMAs, V^T fragments from LDS ----
#pragma unroll
        for (int j = 0; j < 4; ++j) {
            const unsigned short* vp = &vS[bi][0] + (j * 16 + nl) * 32;
            FragBF vf;
            vf.u4[0] = *(const uint4*)(vp + hh * 8);
            vf.u4[1] = *(const uint4*)(vp + 16 + hh * 8);
            o[j] = wmma_bf16(pf.bf, vf.bf, o[j]);
        }

        __syncthreads();   // all reads of buf[bi] done before it is restaged
    }

    // ---- normalize + store ctx[b, s, head*64 + dh] as bf16 ----
    float inv[8];
#pragma unroll
    for (int r = 0; r < 8; ++r) inv[r] = 1.0f / lrow[r];
#pragma unroll
    for (int j = 0; j < 4; ++j) {
#pragma unroll
        for (int r = 0; r < 8; ++r) {
            const int s = st * 16 + hh * 8 + r;
            ctx[((size_t)(b * SS + s)) * DD + head * DH + j * 16 + nl] =
                f2bf(o[j][r] * inv[r]);
        }
    }
}

// ---------------------------------------------------------------------------
// Workspace layout (bytes), total ~88 MB
// ---------------------------------------------------------------------------
#define OFF_XBF   ((size_t)0)
#define OFF_WQKV  ((size_t)16777216)
#define OFF_WFC   ((size_t)23068672)
#define OFF_Q     ((size_t)25165824)
#define OFF_K     ((size_t)41943040)
#define OFF_VT    ((size_t)58720256)
#define OFF_CTX   ((size_t)75497472)

extern "C" void kernel_launch(void* const* d_in, const int* in_sizes, int n_in,
                              void* d_out, int out_size, void* d_ws, size_t ws_size,
                              hipStream_t stream) {
    const float* x     = (const float*)d_in[0];
    const int*   mask  = (const int*)d_in[1];
    // d_in[2] = qmask (unused by the reference)
    const int*   useg  = (const int*)d_in[3];
    const float* wqkv  = (const float*)d_in[4];
    const float* wfc   = (const float*)d_in[5];
    const float* bfc   = (const float*)d_in[6];
    const float* shift = (const float*)d_in[7];
    const float* biasp = (const float*)d_in[8];
    float*       out   = (float*)d_out;

    char* ws = (char*)d_ws;
    unsigned short* xbf    = (unsigned short*)(ws + OFF_XBF);
    unsigned short* wqkvbf = (unsigned short*)(ws + OFF_WQKV);
    unsigned short* wfcbf  = (unsigned short*)(ws + OFF_WFC);
    unsigned short* qb     = (unsigned short*)(ws + OFF_Q);
    unsigned short* kb     = (unsigned short*)(ws + OFF_K);
    unsigned short* vt     = (unsigned short*)(ws + OFF_VT);
    unsigned short* ctx    = (unsigned short*)(ws + OFF_CTX);

    const int nX = BB * SS * DD;       // 8388608
    const int nW = 3 * DD * DD;        // 3145728
    const int nF = DD * DD;            // 1048576

    k_cvt_bf16<<<(nX + 255) / 256, 256, 0, stream>>>(x, xbf, nX);
    k_cvt_bf16<<<(nW + 255) / 256, 256, 0, stream>>>(wqkv, wqkvbf, nW);
    k_cvt_bf16<<<(nF + 255) / 256, 256, 0, stream>>>(wfc, wfcbf, nF);

    k_qkv_gemm<<<dim3(3 * DD / 128, BB * SS / 128), 128, 0, stream>>>(
        xbf, wqkvbf, qb, kb, vt);

    k_attention<<<(BB * HH_ * (SS / 16)) / 8, 256, 0, stream>>>(
        qb, kb, vt, mask, useg, shift, biasp, ctx);

    k_fc_gemm<<<dim3(DD / 128, BB * SS / 128), 128, 0, stream>>>(
        ctx, wfcbf, bfc, out);
}
